// CLEAR_46643344834801
// MI455X (gfx1250) — matrix operand order repaired and checked by reading, not compiled, not run on hardware
//
#include <hip/hip_runtime.h>
#include <stdint.h>

// ---------------------------------------------------------------------------
// Fused SOM soft-assignment kernel for gfx1250 (MI455X), bf16 WMMA path.
//   d_total = cdist(x,P) + cdist(x,G);  w = dbl-normalized exp(-d/T)*gate
//   blended = w @ P
// Memory-bound problem (~200MB traffic -> ~8.6us floor @ 23.3TB/s); bf16
// v_wmma_f32_16x16x32_bf16 keeps the 13 GFLOP of GEMM work far below that.
// ---------------------------------------------------------------------------

typedef __attribute__((ext_vector_type(16))) __bf16          v16bf;
typedef __attribute__((ext_vector_type(16))) unsigned short  u16x16;
typedef __attribute__((ext_vector_type(8)))  float           v8f;

#define N_PTS   131072
#define DIM     64
#define CELLS   256

#define WAVES      4
#define TROWS      16                  // rows (points) per wave tile
#define BLOCK_ROWS (WAVES * TROWS)     // 64 rows per block
#define TPB        (WAVES * 32)        // 128 threads, wave32

// padded LDS strides (elements) chosen for conflict-free 16-lane gathers
#define PROW   (DIM + 2)     // 66  (bf16)  proto/grid row-major
#define PTROW  (CELLS + 2)   // 258 (bf16)  proto transposed
#define XROW   (DIM + 4)     // 68  (bf16)  x tile
#define WROW   (CELLS + 1)   // 257 (f32)   per-wave weight tile

// dynamic LDS layout (bytes)
#define OFF_PROTO   0
#define OFF_GRID    (OFF_PROTO  + CELLS * PROW * 2)
#define OFF_PROTOT  (OFF_GRID   + CELLS * PROW * 2)
#define OFF_PN      (OFF_PROTOT + DIM * PTROW * 2)
#define OFF_GN      (OFF_PN     + CELLS * 4)
#define OFF_GATE    (OFF_GN     + CELLS * 4)
#define OFF_XN      (OFF_GATE   + CELLS * 4)
#define OFF_SCALE   (OFF_XN     + WAVES * TROWS * 4)
#define OFF_WAVE    (OFF_SCALE  + WAVES * TROWS * 4)
#define WAVE_BYTES  (TROWS * XROW * 2 + TROWS * WROW * 4)
#define SMEM_TOTAL  (OFF_WAVE + WAVES * WAVE_BYTES)

__device__ inline unsigned short f2bf(float f) {
  uint32_t u = __builtin_bit_cast(uint32_t, f);
  return (unsigned short)((u + 0x7FFFu + ((u >> 16) & 1u)) >> 16);  // RNE
}
__device__ inline float sigmoidf(float v) { return 1.0f / (1.0f + __expf(-v)); }

__device__ inline v8f wmma_bf16(u16x16 a, u16x16 b, v8f c) {
  return __builtin_amdgcn_wmma_f32_16x16x32_bf16(
      /*neg_a=*/false, __builtin_bit_cast(v16bf, a),
      /*neg_b=*/false, __builtin_bit_cast(v16bf, b),
      /*c_mod=*/(short)0, c, /*reuse_a=*/false, /*reuse_b=*/false);
}

// A fragment (16x32 bf16): lane m=lane&15; halves hold K {0..7,16..23} / {8..15,24..31}
__device__ inline u16x16 loadA_bf(const unsigned short* base, int stride,
                                  int m, int k0, int hf) {
  u16x16 a;
#pragma unroll
  for (int v = 0; v < 8; ++v) {
    int k = k0 + hf * 8 + ((v >> 2) << 4) + ((v & 3) << 1);
    a[2 * v]     = base[m * stride + k];
    a[2 * v + 1] = base[m * stride + k + 1];
  }
  return a;
}
// Same fragment but sourced from f32 LDS with on-the-fly bf16 conversion
__device__ inline u16x16 loadA_f32(const float* base, int stride,
                                   int m, int k0, int hf) {
  u16x16 a;
#pragma unroll
  for (int v = 0; v < 8; ++v) {
    int k = k0 + hf * 8 + ((v >> 2) << 4) + ((v & 3) << 1);
    a[2 * v]     = f2bf(base[m * stride + k]);
    a[2 * v + 1] = f2bf(base[m * stride + k + 1]);
  }
  return a;
}
// B fragment (32x16 bf16): lane n=lane&15; halves hold K {0..7,16..23} / {8..15,24..31}
// base rows are the "N" axis (cells / dims), columns are contiguous K.
__device__ inline u16x16 loadB_bf(const unsigned short* base, int stride,
                                  int n, int k0, int hf) {
  u16x16 b;
#pragma unroll
  for (int v = 0; v < 8; ++v) {
    int k = k0 + hf * 8 + ((v >> 2) << 4) + ((v & 3) << 1);
    b[2 * v]     = base[n * stride + k];
    b[2 * v + 1] = base[n * stride + k + 1];
  }
  return b;
}

extern "C" __global__ __launch_bounds__(TPB, 1)
void SOMFUSE_46643344834801_kernel(const float* __restrict__ x,
                                   const float* __restrict__ proto,
                                   const float* __restrict__ gridp,
                                   const float* __restrict__ gatel,
                                   const float* __restrict__ traw,
                                   float* __restrict__ out_blend,
                                   float* __restrict__ out_w) {
  extern __shared__ __align__(16) char smem[];
  unsigned short* sProto  = (unsigned short*)(smem + OFF_PROTO);
  unsigned short* sGrid   = (unsigned short*)(smem + OFF_GRID);
  unsigned short* sProtoT = (unsigned short*)(smem + OFF_PROTOT);
  float* sPN    = (float*)(smem + OFF_PN);
  float* sGN    = (float*)(smem + OFF_GN);
  float* sGate  = (float*)(smem + OFF_GATE);
  float* sXN    = (float*)(smem + OFF_XN);
  float* sScale = (float*)(smem + OFF_SCALE);

  const int tid  = threadIdx.x;
  const int wave = tid >> 5;
  const int lane = tid & 31;
  const int m    = lane & 15;   // row (A) / col (B,D) index within fragment
  const int hf   = lane >> 4;   // lane half

  unsigned short* sX = (unsigned short*)(smem + OFF_WAVE + wave * WAVE_BYTES);
  float*          sW = (float*)((char*)sX + TROWS * XROW * 2);

  // ---- block init: temperature, gate, bf16 prototypes/grid + norms ----
  const float temp = sigmoidf(traw[0]) * (1.0f - 0.001f) + 0.001f;
  const float nit  = -1.0f / temp;
  for (int c = tid; c < CELLS; c += TPB) {
    sGate[c] = sigmoidf(gatel[c]);
    float pn = 0.f, gn = 0.f;
#pragma unroll 4
    for (int d = 0; d < DIM; ++d) {
      float pv = proto[c * DIM + d];
      float gv = gridp[c * DIM + d];
      pn += pv * pv;
      gn += gv * gv;
      unsigned short pb = f2bf(pv);
      sProto[c * PROW + d]  = pb;
      sProtoT[d * PTROW + c] = pb;
      sGrid[c * PROW + d]   = f2bf(gv);
    }
    sPN[c] = pn;
    sGN[c] = gn;
  }

  // ---- per-wave x tile load (coalesced) + bf16 convert; f32 stashed in sW ----
  const long rowBase = (long)blockIdx.x * BLOCK_ROWS + wave * TROWS;
  for (int i = lane; i < TROWS * DIM; i += 32) {
    int r = i >> 6, d = i & (DIM - 1);
    float v = x[(rowBase + r) * DIM + d];
    sX[r * XROW + d] = f2bf(v);
    sW[r * WROW + d] = v;
  }
  __syncthreads();

  if (lane < TROWS) {
    float s = 0.f;
#pragma unroll 8
    for (int d = 0; d < DIM; ++d) { float v = sW[lane * WROW + d]; s += v * v; }
    sXN[wave * TROWS + lane] = s;
  }
  __syncthreads();

  // ---- distance GEMMs: S1 = X.P^T, S2 = X.G^T  (bf16 WMMA, K=64) ----
  u16x16 a0 = loadA_bf(sX, XROW, m, 0, hf);
  u16x16 a1 = loadA_bf(sX, XROW, m, 32, hf);
  float xnr[8];
#pragma unroll
  for (int r = 0; r < 8; ++r) xnr[r] = sXN[wave * TROWS + r + 8 * hf];

  for (int ct = 0; ct < CELLS / 16; ++ct) {
    const unsigned short* bp = sProto + ct * 16 * PROW;
    const unsigned short* bg = sGrid  + ct * 16 * PROW;
    v8f s1 = {}; v8f s2 = {};
    s1 = wmma_bf16(a0, loadB_bf(bp, PROW, m, 0, hf), s1);
    s1 = wmma_bf16(a1, loadB_bf(bp, PROW, m, 32, hf), s1);
    s2 = wmma_bf16(a0, loadB_bf(bg, PROW, m, 0, hf), s2);
    s2 = wmma_bf16(a1, loadB_bf(bg, PROW, m, 32, hf), s2);

    int cell = ct * 16 + m;
    float pn = sPN[cell], gn = sGN[cell];
#pragma unroll
    for (int r = 0; r < 8; ++r) {
      int mr = r + 8 * hf;                      // D-fragment row
      float d1 = sqrtf(fmaxf(xnr[r] + pn - 2.f * s1[r], 0.f));
      float d2 = sqrtf(fmaxf(xnr[r] + gn - 2.f * s2[r], 0.f));
      sW[mr * WROW + cell] = __expf((d1 + d2) * nit);
    }
  }
  __syncthreads();

  // ---- double normalization folded into one per-row scale ----
  {
    float s = 0.f, t = 0.f;
    for (int j = 0; j < 128; ++j) {
      int c = hf * 128 + j;
      float v = sW[m * WROW + c];
      s += v;
      t += v * sGate[c];
    }
    s += __shfl_xor(s, 16);
    t += __shfl_xor(t, 16);
    if (hf == 0) {
      float S1 = s + 1e-8f;                // sum(e) + eps
      float S2 = t / S1 + 1e-8f;           // sum(e*gate)/S1 + eps
      sScale[wave * TROWS + m] = 1.0f / (S1 * S2);
    }
  }
  __syncthreads();

  // ---- finalize w, write out (coalesced), keep f32 in LDS for W@P ----
  for (int i = lane; i < TROWS * CELLS; i += 32) {
    int r = i >> 8, c = i & (CELLS - 1);
    float v = sW[r * WROW + c] * sGate[c] * sScale[wave * TROWS + r];
    out_w[(rowBase + r) * CELLS + c] = v;
    sW[r * WROW + c] = v;
  }
  __syncthreads();

  // ---- blended = W(16x256) @ P(256x64), bf16 WMMA over K=256 ----
  v8f acc0 = {}, acc1 = {}, acc2 = {}, acc3 = {};
#pragma unroll
  for (int ks = 0; ks < 8; ++ks) {
    u16x16 aw = loadA_f32(sW, WROW, m, ks * 32, hf);
    acc0 = wmma_bf16(aw, loadB_bf(sProtoT + 0 * 16 * PTROW, PTROW, m, ks * 32, hf), acc0);
    acc1 = wmma_bf16(aw, loadB_bf(sProtoT + 1 * 16 * PTROW, PTROW, m, ks * 32, hf), acc1);
    acc2 = wmma_bf16(aw, loadB_bf(sProtoT + 2 * 16 * PTROW, PTROW, m, ks * 32, hf), acc2);
    acc3 = wmma_bf16(aw, loadB_bf(sProtoT + 3 * 16 * PTROW, PTROW, m, ks * 32, hf), acc3);
  }
#pragma unroll
  for (int r = 0; r < 8; ++r) {
    int mr = r + 8 * hf;
    float* o = out_blend + (rowBase + mr) * DIM + m;
    o[0]  = acc0[r];
    o[16] = acc1[r];
    o[32] = acc2[r];
    o[48] = acc3[r];
  }
}

extern "C" void kernel_launch(void* const* d_in, const int* in_sizes, int n_in,
                              void* d_out, int out_size, void* d_ws, size_t ws_size,
                              hipStream_t stream) {
  (void)in_sizes; (void)n_in; (void)d_ws; (void)ws_size; (void)out_size;
  const float* x     = (const float*)d_in[0];
  const float* proto = (const float*)d_in[1];
  const float* gridp = (const float*)d_in[2];
  const float* gatel = (const float*)d_in[3];
  const float* traw  = (const float*)d_in[4];

  float* out_blend = (float*)d_out;                       // [N, 64]
  float* out_w     = out_blend + (size_t)N_PTS * DIM;     // [N, 256]

  (void)hipFuncSetAttribute((const void*)SOMFUSE_46643344834801_kernel,
                            hipFuncAttributeMaxDynamicSharedMemorySize,
                            (int)SMEM_TOTAL);

  dim3 g(N_PTS / BLOCK_ROWS), b(TPB);
  SOMFUSE_46643344834801_kernel<<<g, b, SMEM_TOTAL, stream>>>(
      x, proto, gridp, gatel, traw, out_blend, out_w);
}
